// GeneralRenderingNetwork_32186484916670
// MI455X (gfx1250) — matrix operand order repaired and checked by reading, not compiled, not run on hardware
//
#include <hip/hip_runtime.h>
#include <hip/hip_bf16.h>

// ---------------------------------------------------------------------------
// Fused IBRNet rendering network for gfx1250 (MI455X).
//  prep  : convert all WMMA-layer weights (incl. direction MLP) to f16
//          transposed layout with conflict-avoiding padded kpads, stage small
//          f32 params, zero per-ray counters.
//  main  : one block = 16 samples of one ray = 128 rows (row = sample*8+view).
//          rgb_feat staged coalesced via global_load_async_to_lds_b32.
//          rd1(4->16), rd2(16->59), b1(185->64), b2(64->32), v1, v2(->33),
//          u1, r1(37->16) all via v_wmma_f32_16x16x32_f16; u2/r2/r3 scalar;
//          per-sample softmax over 8 views -> rgb_out.
//          LDS strides padded (200/72/40) so fragment-load lanes spread over
//          16 bank positions instead of 2 (64-bank LDS, b128 accesses).
//  valid : per-ray valid_mask from counters.
// ---------------------------------------------------------------------------

typedef __attribute__((ext_vector_type(16))) _Float16 v16h;
typedef __attribute__((ext_vector_type(8)))  _Float16 v8h;
typedef __attribute__((ext_vector_type(8)))  float    v8f;

// padded strides (halves; multiples of 8 for 16B-aligned b128 loads,
// dword-stride gcd with 64 banks == 4)
#define XST   200   // sX row stride (192 cols used)
#define KB1   200
#define KB2   72
#define KV1   40
#define KV2   40
#define KU1   40
#define KR1   72
#define KRD1  40
#define KRD2  40

// ---- workspace layout -------------------------------------------------------
// [0, 4096)                : 1024 float per-ray counters
// [4096, 4096+2*23936)     : f16 transposed weight arena
// [51968, 51968+4*478)     : f32 small-param arena
#define WS_WT_OFF   4096
#define WS_SP_OFF   51968

// f16 arena offsets (halves)
#define H_B1  0        // 64 x KB1  (k 185.. zero)
#define H_B2  12800    // 32 x KB2  (k 64.. zero)
#define H_V1  15104    // 32 x KV1  (k 32.. zero)
#define H_V2  16384    // 48 x KV2  (n 33..47, k 32.. zero)
#define H_U1  18304    // 32 x KU1
#define H_R1  19584    // 16 x KR1  (k 37.. zero)
#define H_RD1 20736    // 16 x KRD1 (k 4.. zero)
#define H_RD2 21376    // 64 x KRD2 (n 59..63, k 16.. zero)
#define H_TOTAL 23936

// f32 small arena offsets (floats)
#define OFF_BRD1 0      // 16
#define OFF_BRD2 16     // 59
#define OFF_BB1  75     // 64
#define OFF_BB2  139    // 32
#define OFF_BV1  171    // 32
#define OFF_BV2  203    // 48 (padded)
#define OFF_BU1  251    // 32
#define OFF_WU2  283    // 32
#define OFF_BU2  315    // 1
#define OFF_BR1  316    // 16
#define OFF_WR2  332    // 16x8
#define OFF_BR2  460    // 8
#define OFF_WR3  468    // 8
#define OFF_BR3  476    // 1
#define OFF_S    477    // 1
#define SP_TOTAL 478

struct PW {
  const float *w_rd1,*b_rd1,*w_rd2,*b_rd2,*w_b1,*b_b1,*w_b2,*b_b2;
  const float *w_v1,*b_v1,*w_v2,*b_v2,*w_u1,*b_u1,*w_u2,*b_u2;
  const float *w_r1,*b_r1,*w_r2,*b_r2,*w_r3,*b_r3,*s;
};

__device__ __forceinline__ float eluf(float x){ return x > 0.f ? x : __expf(x) - 1.f; }
__device__ __forceinline__ float sigm(float x){ return 1.f / (1.f + __expf(-x)); }

// A fragment: 16x32 f16, rows r0..r0+15 from LDS row-major (strideH halves).
// Lane L<16: M=L, K = {k0..k0+7, k0+16..k0+23}; lanes 16..31: K = {+8, +24}.
__device__ __forceinline__ v16h ldsA(const _Float16* base, int r0, int strideH,
                                     int k0, int lane) {
  int lr = lane & 15, hf = lane >> 4;
  const _Float16* p = base + (size_t)(r0 + lr) * strideH + k0 + hf * 8;
  v8h lo = *(const v8h*)p;
  v8h hi = *(const v8h*)(p + 16);
  return __builtin_shufflevector(lo, hi, 0,1,2,3,4,5,6,7,8,9,10,11,12,13,14,15);
}

// B fragment: 32x16 f16 from transposed weights Wt[N][kpad] (row-major in N).
// Lane L<16: N=L, K = {k0..k0+7, k0+16..k0+23}; lanes 16..31: N=L-16, K = {+8, +24}.
__device__ __forceinline__ v16h ldsB(const _Float16* wt, int n0, int kpad,
                                     int k0, int lane) {
  int lr = lane & 15, hf = lane >> 4;
  const _Float16* p = wt + (size_t)(n0 + lr) * kpad + k0 + hf * 8;
  v8h lo = *(const v8h*)p;
  v8h hi = *(const v8h*)(p + 16);
  return __builtin_shufflevector(lo, hi, 0,1,2,3,4,5,6,7,8,9,10,11,12,13,14,15);
}

__device__ __forceinline__ v8f wmma16(v16h a, v16h b, v8f c) {
  return __builtin_amdgcn_wmma_f32_16x16x32_f16(false, a, false, b, (short)0, c,
                                                false, false);
}

// Async global -> LDS copy of one dword per lane (CDNA5 GLOBAL_LOAD_ASYNC_TO_LDS_B32).
__device__ __forceinline__ void async_g2l_b32(unsigned lds_off,
                                              unsigned long long gaddr) {
  asm volatile("global_load_async_to_lds_b32 %0, %1, off"
               :: "v"(lds_off), "v"(gaddr) : "memory");
}
__device__ __forceinline__ void wait_asynccnt0() {
  asm volatile("s_wait_asynccnt 0" ::: "memory");
}

// ---------------------------------------------------------------------------
__global__ void prep_kernel(PW p, float* ws_f) {
  char* ws = (char*)ws_f;
  float* counters = (float*)ws;
  _Float16* wt = (_Float16*)(ws + WS_WT_OFF);
  float* sp = (float*)(ws + WS_SP_OFF);
  int t = threadIdx.x;

  for (int i = t; i < 1024; i += 256) counters[i] = 0.f;

  for (int i = t; i < 64*KB1; i += 256) { int n=i/KB1, k=i%KB1;
    wt[H_B1+i] = (k<185) ? (_Float16)p.w_b1[k*64+n] : (_Float16)0.f; }
  for (int i = t; i < 32*KB2; i += 256) { int n=i/KB2, k=i%KB2;
    wt[H_B2+i] = (k<64) ? (_Float16)p.w_b2[k*32+n] : (_Float16)0.f; }
  for (int i = t; i < 32*KV1; i += 256) { int n=i/KV1, k=i%KV1;
    wt[H_V1+i] = (k<32) ? (_Float16)p.w_v1[k*32+n] : (_Float16)0.f; }
  for (int i = t; i < 48*KV2; i += 256) { int n=i/KV2, k=i%KV2;
    wt[H_V2+i] = (k<32 && n<33) ? (_Float16)p.w_v2[k*33+n] : (_Float16)0.f; }
  for (int i = t; i < 32*KU1; i += 256) { int n=i/KU1, k=i%KU1;
    wt[H_U1+i] = (k<32) ? (_Float16)p.w_u1[k*32+n] : (_Float16)0.f; }
  for (int i = t; i < 16*KR1; i += 256) { int n=i/KR1, k=i%KR1;
    wt[H_R1+i] = (k<37) ? (_Float16)p.w_r1[k*16+n] : (_Float16)0.f; }
  for (int i = t; i < 16*KRD1; i += 256) { int n=i/KRD1, k=i%KRD1;
    wt[H_RD1+i] = (k<4) ? (_Float16)p.w_rd1[k*16+n] : (_Float16)0.f; }
  for (int i = t; i < 64*KRD2; i += 256) { int n=i/KRD2, k=i%KRD2;
    wt[H_RD2+i] = (k<16 && n<59) ? (_Float16)p.w_rd2[k*59+n] : (_Float16)0.f; }

  for (int i = t; i < 16;  i += 256) sp[OFF_BRD1+i] = p.b_rd1[i];
  for (int i = t; i < 59;  i += 256) sp[OFF_BRD2+i] = p.b_rd2[i];
  for (int i = t; i < 64;  i += 256) sp[OFF_BB1+i]  = p.b_b1[i];
  for (int i = t; i < 32;  i += 256) sp[OFF_BB2+i]  = p.b_b2[i];
  for (int i = t; i < 32;  i += 256) sp[OFF_BV1+i]  = p.b_v1[i];
  for (int i = t; i < 48;  i += 256) sp[OFF_BV2+i]  = (i<33) ? p.b_v2[i] : 0.f;
  for (int i = t; i < 32;  i += 256) sp[OFF_BU1+i]  = p.b_u1[i];
  for (int i = t; i < 32;  i += 256) sp[OFF_WU2+i]  = p.w_u2[i];
  for (int i = t; i < 16;  i += 256) sp[OFF_BR1+i]  = p.b_r1[i];
  for (int i = t; i < 128; i += 256) sp[OFF_WR2+i]  = p.w_r2[i];
  for (int i = t; i < 8;   i += 256) sp[OFF_BR2+i]  = p.b_r2[i];
  for (int i = t; i < 8;   i += 256) sp[OFF_WR3+i]  = p.w_r3[i];
  if (t == 0) { sp[OFF_BU2] = p.b_u2[0]; sp[OFF_BR3] = p.b_r3[0]; sp[OFF_S] = p.s[0]; }
}

// ---------------------------------------------------------------------------
__global__ __launch_bounds__(256) void render_main(
    const float* __restrict__ geom_g, const float* __restrict__ rgb_g,
    const float* __restrict__ rayd_g, const float* __restrict__ mask_g,
    float* __restrict__ ws_f, float* __restrict__ out) {
  char* ws = (char*)ws_f;
  float* counters = (float*)ws;
  const _Float16* wt_g = (const _Float16*)(ws + WS_WT_OFF);
  const float* sp_g = (const float*)(ws + WS_SP_OFF);

  __shared__ __align__(16) _Float16 sX[128 * XST];      // b1 input
  __shared__ __align__(16) _Float16 sS[128 * 64 * 2];   // s1 | s2 ping-pong (+f32 alias)
  __shared__ __align__(16) _Float16 sX32[128 * 32];     // rd-input, later x (post b2)
  __shared__ __align__(16) _Float16 sXN[128 * 32];      // rd-hidden / mean-var / x+x_res
  __shared__ __align__(16) _Float16 sW[H_TOTAL];
  __shared__ float sP[SP_TOTAL];
  __shared__ float sRD[128 * 4];
  __shared__ float sMask[128];
  __shared__ float sRGBin[128 * 3];
  __shared__ float sExpD[128];
  __shared__ float sWgt[128];
  __shared__ float sVis[128];
  __shared__ float sScore[128];
  __shared__ float sMin[16], sSum[16], sFlag[16];

  const int tid = threadIdx.x, lane = tid & 31, wave = tid >> 5;
  const int ray = blockIdx.x >> 3, s0 = (blockIdx.x & 7) * 16;
  const int R0 = wave * 16;
  _Float16* s1 = sS;
  _Float16* s2 = sS + 8192;
  float* rgbf = (float*)sS;     // f32 alias: raw/updated rgb_feat [128][59]
  float* sMV  = (float*)sXN;    // f32 alias (7552 B <= 8192): mean|var, phases C/D only

  // ---- Phase A: async-stage rgb_feat; stage weights/params/rd/mask/geom -----
  // rgb_feat: per view, (16 samples x 59 feats) is contiguous -> coalesced.
  for (int i = tid; i < 128 * 59; i += 256) {
    int v = i / 944, rem = i % 944, s = rem / 59, j = rem % 59;
    const float* ga = rgb_g + (((size_t)v * 1024 + ray) * 128 + (s0 + s)) * 59 + j;
    unsigned lds = (unsigned)(uintptr_t)(rgbf + ((s * 8 + v) * 59 + j));
    async_g2l_b32(lds, (unsigned long long)(uintptr_t)ga);
  }
  for (int i = tid; i < H_TOTAL;  i += 256) sW[i] = wt_g[i];
  for (int i = tid; i < SP_TOTAL; i += 256) sP[i] = sp_g[i];
  // ray_diff -> sRD (f32) and rd-input (f16, K padded 4->32) in sX32
  for (int i = tid; i < 128 * 32; i += 256) {
    int lr = i >> 5, c = i & 31;
    if (c < 4) {
      int s = s0 + (lr >> 3), v = lr & 7;
      float val = rayd_g[(((size_t)v * 1024 + ray) * 128 + s) * 4 + c];
      sRD[lr*4 + c] = val;
      sX32[i] = (_Float16)val;
      if (c == 3) sExpD[lr] = __expf(fabsf(sp_g[OFF_S]) * (val - 1.f));
    } else sX32[i] = (_Float16)0.f;
  }
  for (int i = tid; i < 128 * 32; i += 256) sXN[i] = (_Float16)0.f; // rd-hidden K pad
  for (int i = tid; i < 128; i += 256) { int s = s0 + (i >> 3), v = i & 7;
    sMask[i] = mask_g[((size_t)v * 1024 + ray) * 128 + s]; }
  for (int i = tid; i < 128 * 8; i += 256) { int row = i >> 3, c = i & 7;
    int s = s0 + (row >> 3);
    sX[row*XST + c] = (_Float16)geom_g[((size_t)ray * 128 + s) * 8 + c]; }
  for (int i = tid; i < 128 * 15; i += 256) { int row = i / 15, k = 185 + i % 15;
    sX[row*XST + k] = (_Float16)0.f; }
  wait_asynccnt0();
  __syncthreads();

  const int lr = lane & 15, hf = lane >> 4;

  // ---- snapshot rgb_in (pre-direction) + Phase B1: rd1 (K=32 -> N=16) -------
  for (int i = tid; i < 384; i += 256) { int row = i / 3, c = i % 3;
    sRGBin[i] = rgbf[row*59 + c]; }
  {
    v8f acc = {0,0,0,0,0,0,0,0};
    v16h a = ldsA(sX32, R0, 32, 0, lane);
    v16h b = ldsB(sW + H_RD1, 0, KRD1, 0, lane);
    acc = wmma16(a, b, acc);
    float bias = sP[OFF_BRD1 + lr];
#pragma unroll
    for (int r = 0; r < 8; r++) { int row = R0 + r + hf*8;
      sXN[row*32 + lr] = (_Float16)eluf(acc[r] + bias); }
  }
  __syncthreads();

  // ---- Phase B2: rd2 (K=32 -> N=64, 59 valid); rgb_feat += elu(dir) ---------
  {
    v8f acc[4] = {{0,0,0,0,0,0,0,0},{0,0,0,0,0,0,0,0},
                  {0,0,0,0,0,0,0,0},{0,0,0,0,0,0,0,0}};
    v16h a = ldsA(sXN, R0, 32, 0, lane);
#pragma unroll
    for (int nt = 0; nt < 4; nt++) {
      v16h b = ldsB(sW + H_RD2, nt*16, KRD2, 0, lane);
      acc[nt] = wmma16(a, b, acc[nt]);
    }
#pragma unroll
    for (int nt = 0; nt < 4; nt++) {
      int j = nt*16 + lr;
      if (j < 59) {
        float bias = sP[OFF_BRD2 + j];
#pragma unroll
        for (int r = 0; r < 8; r++) { int row = R0 + r + hf*8;
          float rr = rgbf[row*59 + j] + eluf(acc[nt][r] + bias);
          rgbf[row*59 + j] = rr;
          sX[row*XST + 126 + j] = (_Float16)rr; }
      }
    }
  }
  __syncthreads();

  // ---- Phase C: per-sample view weights, weighted mean/var ------------------
  if (tid < 16) {
    float mn = 1e30f;
#pragma unroll
    for (int v = 0; v < 8; v++) mn = fminf(mn, sExpD[tid*8+v]);
    float sm = 0.f;
#pragma unroll
    for (int v = 0; v < 8; v++) sm += (sExpD[tid*8+v] - mn) * sMask[tid*8+v];
    sMin[tid] = mn; sSum[tid] = sm;
  }
  __syncthreads();
  if (tid < 128) { int s = tid >> 3;
    sWgt[tid] = (sExpD[tid] - sMin[s]) * sMask[tid] / (sSum[s] + 1e-8f); }
  __syncthreads();
  for (int i = tid; i < 944; i += 256) {
    int sl = i / 59, j = i % 59;
    float m = 0.f;
#pragma unroll
    for (int v = 0; v < 8; v++) m += sWgt[sl*8+v] * rgbf[(sl*8+v)*59+j];
    float vv = 0.f;
#pragma unroll
    for (int v = 0; v < 8; v++) { float d = rgbf[(sl*8+v)*59+j] - m;
      vv += sWgt[sl*8+v] * d * d; }
    sMV[i] = m; sMV[944 + i] = vv;
  }
  __syncthreads();
  if (tid < 128) { int s = tid >> 3;
    for (int j = 0; j < 59; j++) sX[tid*XST + 8  + j] = (_Float16)sMV[s*59+j];
    for (int j = 0; j < 59; j++) sX[tid*XST + 67 + j] = (_Float16)sMV[944 + s*59+j];
  }
  __syncthreads();

  // ---- Phase E: b1 (K=192 -> N=64), bias+ELU -> s1[128][64] -----------------
  {
    v8f acc[4] = {{0,0,0,0,0,0,0,0},{0,0,0,0,0,0,0,0},
                  {0,0,0,0,0,0,0,0},{0,0,0,0,0,0,0,0}};
    for (int kt = 0; kt < 6; kt++) {
      v16h a = ldsA(sX, R0, XST, kt*32, lane);
#pragma unroll
      for (int nt = 0; nt < 4; nt++) {
        v16h b = ldsB(sW + H_B1, nt*16, KB1, kt*32, lane);
        acc[nt] = wmma16(a, b, acc[nt]);
      }
    }
#pragma unroll
    for (int nt = 0; nt < 4; nt++) {
      float bias = sP[OFF_BB1 + nt*16 + lr];
#pragma unroll
      for (int r = 0; r < 8; r++) { int row = R0 + r + hf*8;
        s1[row*64 + nt*16 + lr] = (_Float16)eluf(acc[nt][r] + bias); }
    }
  }
  __syncthreads();

  // ---- Phase F: b2 (K=64 -> N=32) -> x (sX32) and x*weight (s2) -------------
  {
    v8f acc[2] = {{0,0,0,0,0,0,0,0},{0,0,0,0,0,0,0,0}};
    for (int kt = 0; kt < 2; kt++) {
      v16h a = ldsA(s1, R0, 64, kt*32, lane);
#pragma unroll
      for (int nt = 0; nt < 2; nt++) {
        v16h b = ldsB(sW + H_B2, nt*16, KB2, kt*32, lane);
        acc[nt] = wmma16(a, b, acc[nt]);
      }
    }
#pragma unroll
    for (int nt = 0; nt < 2; nt++) {
      float bias = sP[OFF_BB2 + nt*16 + lr];
#pragma unroll
      for (int r = 0; r < 8; r++) { int row = R0 + r + hf*8;
        float x = eluf(acc[nt][r] + bias);
        sX32[row*32 + nt*16 + lr] = (_Float16)x;
        s2  [row*32 + nt*16 + lr] = (_Float16)(x * sWgt[row]); }
    }
  }
  __syncthreads();

  // ---- Phase G: v1 (K=32 -> N=32), ELU -> s1 (stride 32) --------------------
  {
    v8f acc[2] = {{0,0,0,0,0,0,0,0},{0,0,0,0,0,0,0,0}};
    v16h a = ldsA(s2, R0, 32, 0, lane);
#pragma unroll
    for (int nt = 0; nt < 2; nt++) {
      v16h b = ldsB(sW + H_V1, nt*16, KV1, 0, lane);
      acc[nt] = wmma16(a, b, acc[nt]);
    }
#pragma unroll
    for (int nt = 0; nt < 2; nt++) {
      float bias = sP[OFF_BV1 + nt*16 + lr];
#pragma unroll
      for (int r = 0; r < 8; r++) { int row = R0 + r + hf*8;
        s1[row*32 + nt*16 + lr] = (_Float16)eluf(acc[nt][r] + bias); }
    }
  }

  // ---- Phase H: v2 (K=32 -> N=33): x_res -> xN = x + x_res, vis -------------
  {
    v8f a0 = {0,0,0,0,0,0,0,0}, a1 = {0,0,0,0,0,0,0,0}, a2 = {0,0,0,0,0,0,0,0};
    v16h a  = ldsA(s1, R0, 32, 0, lane);
    v16h b0 = ldsB(sW + H_V2,  0, KV2, 0, lane);
    v16h b1 = ldsB(sW + H_V2, 16, KV2, 0, lane);
    v16h b2 = ldsB(sW + H_V2, 32, KV2, 0, lane);
    a0 = wmma16(a, b0, a0); a1 = wmma16(a, b1, a1); a2 = wmma16(a, b2, a2);
#pragma unroll
    for (int nt = 0; nt < 2; nt++) {
      v8f& ac = nt ? a1 : a0;
      float bias = sP[OFF_BV2 + nt*16 + lr];
#pragma unroll
      for (int r = 0; r < 8; r++) { int row = R0 + r + hf*8;
        float xres = eluf(ac[r] + bias);
        float xn = (float)sX32[row*32 + nt*16 + lr] + xres;
        sXN[row*32 + nt*16 + lr] = (_Float16)xn; }
    }
    if (lr == 0) {
      float bias = sP[OFF_BV2 + 32];
#pragma unroll
      for (int r = 0; r < 8; r++) { int row = R0 + r + hf*8;
        sVis[row] = sigm(eluf(a2[r] + bias)) * sMask[row]; }
    }
  }
  __syncthreads();

  // ---- Phase I: xu = xN * vis -> s2 ----------------------------------------
  if (tid < 128) {
    float v = sVis[tid];
    for (int j = 0; j < 32; j++)
      s2[tid*32 + j] = (_Float16)((float)sXN[tid*32 + j] * v);
  }
  __syncthreads();

  // ---- Phase J1: u1 (K=32 -> N=32), ELU -> s1 ------------------------------
  {
    v8f acc[2] = {{0,0,0,0,0,0,0,0},{0,0,0,0,0,0,0,0}};
    v16h a = ldsA(s2, R0, 32, 0, lane);
#pragma unroll
    for (int nt = 0; nt < 2; nt++) {
      v16h b = ldsB(sW + H_U1, nt*16, KU1, 0, lane);
      acc[nt] = wmma16(a, b, acc[nt]);
    }
#pragma unroll
    for (int nt = 0; nt < 2; nt++) {
      float bias = sP[OFF_BU1 + nt*16 + lr];
#pragma unroll
      for (int r = 0; r < 8; r++) { int row = R0 + r + hf*8;
        s1[row*32 + nt*16 + lr] = (_Float16)eluf(acc[nt][r] + bias); }
    }
  }
  __syncthreads();

  // ---- Phase J2: u2 (32->1) sigmoid*mask; build r-input (37 -> pad 64) ------
  if (tid < 128) {
    float z = sP[OFF_BU2];
    for (int j = 0; j < 32; j++) z += (float)s1[tid*32 + j] * sP[OFF_WU2 + j];
    float vis2 = sigm(z) * sMask[tid];
    for (int j = 0; j < 32; j++) s2[tid*64 + j] = sXN[tid*32 + j];
    s2[tid*64 + 32] = (_Float16)vis2;
    for (int c = 0; c < 4; c++) s2[tid*64 + 33 + c] = (_Float16)sRD[tid*4 + c];
    for (int k = 37; k < 64; k++) s2[tid*64 + k] = (_Float16)0.f;
  }
  __syncthreads();

  // ---- Phase K: r1 (K=64 -> N=16), ELU -> s1 (stride 16) --------------------
  {
    v8f acc = {0,0,0,0,0,0,0,0};
    for (int kt = 0; kt < 2; kt++) {
      v16h a = ldsA(s2, R0, 64, kt*32, lane);
      v16h b = ldsB(sW + H_R1, 0, KR1, kt*32, lane);
      acc = wmma16(a, b, acc);
    }
    float bias = sP[OFF_BR1 + lr];
#pragma unroll
    for (int r = 0; r < 8; r++) { int row = R0 + r + hf*8;
      s1[row*16 + lr] = (_Float16)eluf(acc[r] + bias); }
  }
  __syncthreads();

  // ---- Phase L: r2 (16->8) ELU, r3 (8->1), mask -> score --------------------
  if (tid < 128) {
    float h8[8];
#pragma unroll
    for (int o = 0; o < 8; o++) {
      float z = sP[OFF_BR2 + o];
      for (int k = 0; k < 16; k++) z += (float)s1[tid*16 + k] * sP[OFF_WR2 + k*8 + o];
      h8[o] = eluf(z);
    }
    float z = sP[OFF_BR3];
#pragma unroll
    for (int o = 0; o < 8; o++) z += h8[o] * sP[OFF_WR3 + o];
    sScore[tid] = (sMask[tid] == 0.f) ? -1000000000.f : z;
  }
  __syncthreads();

  // ---- Phase M: per-sample softmax over views, blend rgb, validity ----------
  if (tid < 16) {
    int base = tid * 8;
    float mx = -1e30f;
#pragma unroll
    for (int v = 0; v < 8; v++) mx = fmaxf(mx, sScore[base + v]);
    float e[8]; float den = 0.f;
#pragma unroll
    for (int v = 0; v < 8; v++) { e[v] = __expf(sScore[base + v] - mx); den += e[v]; }
    float inv = 1.f / den;
    int sg = s0 + tid;
#pragma unroll
    for (int c = 0; c < 3; c++) {
      float acc = 0.f;
#pragma unroll
      for (int v = 0; v < 8; v++) acc += e[v] * inv * sRGBin[(base + v)*3 + c];
      out[((size_t)ray * 128 + sg) * 3 + c] = acc;
    }
    float mv = 0.f;
#pragma unroll
    for (int v = 0; v < 8; v++) mv += sMask[base + v];
    sFlag[tid] = (mv >= 2.f) ? 1.f : 0.f;
  }
  __syncthreads();
  if (tid == 0) {
    float c = 0.f;
    for (int i = 0; i < 16; i++) c += sFlag[i];
    atomicAdd(&counters[ray], c);
  }
}

// ---------------------------------------------------------------------------
__global__ void valid_kernel(const float* ws_f, float* out) {
  int t = blockIdx.x * 256 + threadIdx.x;
  if (t < 1024) out[1024 * 128 * 3 + t] = (ws_f[t] > 8.f) ? 1.f : 0.f;
}

// ---------------------------------------------------------------------------
extern "C" void kernel_launch(void* const* d_in, const int* in_sizes, int n_in,
                              void* d_out, int out_size, void* d_ws, size_t ws_size,
                              hipStream_t stream) {
  (void)in_sizes; (void)n_in; (void)out_size; (void)ws_size;
  const float* geom = (const float*)d_in[0];
  const float* rgb  = (const float*)d_in[1];
  const float* rayd = (const float*)d_in[2];
  const float* mask = (const float*)d_in[3];
  PW p;
  p.w_rd1 = (const float*)d_in[4];  p.b_rd1 = (const float*)d_in[5];
  p.w_rd2 = (const float*)d_in[6];  p.b_rd2 = (const float*)d_in[7];
  p.w_b1  = (const float*)d_in[8];  p.b_b1  = (const float*)d_in[9];
  p.w_b2  = (const float*)d_in[10]; p.b_b2  = (const float*)d_in[11];
  p.w_v1  = (const float*)d_in[12]; p.b_v1  = (const float*)d_in[13];
  p.w_v2  = (const float*)d_in[14]; p.b_v2  = (const float*)d_in[15];
  p.w_u1  = (const float*)d_in[16]; p.b_u1  = (const float*)d_in[17];
  p.w_u2  = (const float*)d_in[18]; p.b_u2  = (const float*)d_in[19];
  p.w_r1  = (const float*)d_in[20]; p.b_r1  = (const float*)d_in[21];
  p.w_r2  = (const float*)d_in[22]; p.b_r2  = (const float*)d_in[23];
  p.w_r3  = (const float*)d_in[24]; p.b_r3  = (const float*)d_in[25];
  p.s     = (const float*)d_in[26];
  float* out = (float*)d_out;
  float* ws  = (float*)d_ws;

  prep_kernel<<<1, 256, 0, stream>>>(p, ws);
  render_main<<<8192, 256, 0, stream>>>(geom, rgb, rayd, mask, ws, out);
  valid_kernel<<<4, 256, 0, stream>>>(ws, out);
}
